// Decoder_30253749633586
// MI455X (gfx1250) — matrix operand order repaired and checked by reading, not compiled, not run on hardware
//
#include <hip/hip_runtime.h>
#include <stdint.h>

#define B_SZ 16384
#define T_SZ 24
#define K1   1056      // zx width (1024 + 32), divisible by 32
#define H_   128
#define NGI  384

typedef __attribute__((ext_vector_type(16))) __bf16 v16bf;
typedef __attribute__((ext_vector_type(8)))  float  v8f;

union FragU { v16bf v; uint4 q[2]; };

__device__ __forceinline__ unsigned short f2bf(float f) {
  unsigned u = __float_as_uint(f);
  u += 0x7fffu + ((u >> 16) & 1u);          // round-to-nearest-even
  return (unsigned short)(u >> 16);
}
__device__ __forceinline__ float bf2f(unsigned short h) {
  return __uint_as_float(((unsigned)h) << 16);
}

// CDNA5 has a native V_TANH_F32 transcendental; use it when the builtin exists.
__device__ __forceinline__ float fast_tanh(float x) {
#if __has_builtin(__builtin_amdgcn_tanhf)
  return __builtin_amdgcn_tanhf(x);
#elif __has_builtin(__builtin_amdgcn_tanh_f32)
  return __builtin_amdgcn_tanh_f32(x);
#else
  float e2 = __expf(2.0f * x);
  return (e2 - 1.0f) / (e2 + 1.0f);
#endif
}
__device__ __forceinline__ float fast_sigmoid(float x) {
  return 0.5f + 0.5f * fast_tanh(0.5f * x);
}

// ---------------- conversion / prep kernels ----------------

__global__ void k_conv_zx(const float* __restrict__ enc, const float* __restrict__ z,
                          unsigned short* __restrict__ zx, int total) {
  int i = blockIdx.x * blockDim.x + threadIdx.x;
  if (i >= total) return;
  int row = i / K1, c = i - row * K1;
  float v = (c < 1024) ? enc[(size_t)row * 1024 + c] : z[(size_t)row * 32 + (c - 1024)];
  zx[i] = f2bf(v);
}

// Builds: Wcat (512 x 1056 bf16) = [W_dh ; W_ih[:, :1056] gate-interleaved],
//         Whh  (384 x 128 bf16, gate-interleaved rows),
//         Wa   (384 x 2 f32, per-col a-weights), Wm (4 x 128 f32: mu0,mu1,std0,std1)
__global__ void k_conv_w(const float* __restrict__ W_dh, const float* __restrict__ W_ih,
                         const float* __restrict__ W_hh, const float* __restrict__ W_mu,
                         const float* __restrict__ W_std,
                         unsigned short* __restrict__ Wcat, unsigned short* __restrict__ Whh,
                         float* __restrict__ Wa, float* __restrict__ Wm) {
  const int nWcat = 512 * K1, nWhh = NGI * H_, nWa = NGI * 2, nWm = 4 * H_;
  int i = blockIdx.x * blockDim.x + threadIdx.x;
  if (i < nWcat) {
    int row = i / K1, k = i - row * K1;
    float v;
    if (row < 128) v = W_dh[(size_t)row * K1 + k];
    else {
      int c = row - 128, b = c / 48, w = c % 48, g = w / 16, j = 16 * b + (w & 15);
      v = W_ih[(size_t)(g * 128 + j) * 1060 + k];
    }
    Wcat[i] = f2bf(v);
  } else if (i < nWcat + nWhh) {
    int ii = i - nWcat, c = ii / H_, k = ii - c * H_;
    int b = c / 48, w = c % 48, g = w / 16, j = 16 * b + (w & 15);
    Whh[ii] = f2bf(W_hh[(size_t)(g * 128 + j) * H_ + k]);
  } else if (i < nWcat + nWhh + nWa) {
    int ii = i - nWcat - nWhh, c = ii >> 1, s = ii & 1;
    int b = c / 48, w = c % 48, g = w / 16, j = 16 * b + (w & 15);
    Wa[ii] = W_ih[(size_t)(g * 128 + j) * 1060 + 1056 + s];
  } else if (i < nWcat + nWhh + nWa + nWm) {
    int ii = i - nWcat - nWhh - nWa, o = ii / H_, k = ii - o * H_;
    Wm[ii] = (o < 2) ? W_mu[o * H_ + k] : W_std[(o - 2) * H_ + k];
  }
}

__global__ void k_conv_ra(const float* __restrict__ last, const float* __restrict__ sg,
                          const float* __restrict__ W_vel, const float* __restrict__ b_vel,
                          float* __restrict__ rel, float* __restrict__ a0) {
  int r = blockIdx.x * blockDim.x + threadIdx.x;
  if (r >= B_SZ) return;
  const float inv_dt = 1.0f / (0.4f * 12.0f);
  rel[r * 2 + 0] = (sg[r * 2 + 0] - last[r * 6 + 0]) * inv_dt;
  rel[r * 2 + 1] = (sg[r * 2 + 1] - last[r * 6 + 1]) * inv_dt;
  #pragma unroll
  for (int j = 0; j < 2; ++j) {
    float s = b_vel[j];
    #pragma unroll
    for (int k = 0; k < 6; ++k) s += last[r * 6 + k] * W_vel[j * 6 + k];
    a0[r * 2 + j] = s;
  }
}

// ---------------- big GEMM: [h0 | gi_base] = zx @ Wcat.T ----------------
// grid (128, 4), block 256 (8 wave32s: 4 along M x 2 along N). Tile 128x128.
__global__ __launch_bounds__(256) void k_gemm_pre(
    const unsigned short* __restrict__ zx, const unsigned short* __restrict__ Wcat,
    const float* __restrict__ b_dh, const float* __restrict__ b_ih,
    const float* __restrict__ W_ih, const float* __restrict__ rel,
    float* __restrict__ h0, float* __restrict__ gi) {
  int lane = threadIdx.x & 31, wave = threadIdx.x >> 5;
  int mw = wave & 3, nw = wave >> 2;
  int mrow = blockIdx.x * 128 + mw * 32;        // 32 rows / wave (2 frags)
  int ncol = blockIdx.y * 128 + nw * 64;        // 64 cols / wave (4 frags)
  int l15 = lane & 15, hi = lane >> 4;

  v8f acc[2][4];
  #pragma unroll
  for (int i = 0; i < 2; ++i)
    #pragma unroll
    for (int j = 0; j < 4; ++j)
      #pragma unroll
      for (int d = 0; d < 8; ++d) acc[i][j][d] = 0.0f;

  const unsigned short* Abase = zx   + (size_t)(mrow + l15) * K1 + hi * 8;
  const unsigned short* Bbase = Wcat + (size_t)(ncol + l15) * K1 + hi * 8;

  for (int k = 0; k < K1; k += 32) {
    FragU a[2], b[4];
    #pragma unroll
    for (int i = 0; i < 2; ++i) {               // A: K chunks {0-7,16-23}/{8-15,24-31}
      const uint4* p = (const uint4*)(Abase + (size_t)i * 16 * K1 + k);
      a[i].q[0] = p[0]; a[i].q[1] = p[2];
    }
    #pragma unroll
    for (int j = 0; j < 4; ++j) {               // B: K chunks {0-7,16-23}/{8-15,24-31}
      const uint4* p = (const uint4*)(Bbase + (size_t)j * 16 * K1 + k);
      b[j].q[0] = p[0]; b[j].q[1] = p[2];
    }
    #pragma unroll
    for (int i = 0; i < 2; ++i)
      #pragma unroll
      for (int j = 0; j < 4; ++j)
        acc[i][j] = __builtin_amdgcn_wmma_f32_16x16x32_bf16(
            false, a[i].v, false, b[j].v, (short)0, acc[i][j], false, false);
  }

  #pragma unroll
  for (int i = 0; i < 2; ++i) {
    #pragma unroll
    for (int j = 0; j < 4; ++j) {
      int colv = ncol + j * 16 + l15;
      #pragma unroll
      for (int d = 0; d < 8; ++d) {
        int row = mrow + i * 16 + hi * 8 + d;   // C/D layout: M = d (+8 for hi lanes)
        float v = acc[i][j][d];
        if (colv < 128) {
          h0[(size_t)row * H_ + colv] = v + b_dh[colv];
        } else {
          int c = colv - 128;
          int bb = c / 48, w2 = c % 48, g = w2 / 16, jj = 16 * bb + (w2 & 15);
          int orig = g * 128 + jj;
          v += b_ih[orig]
             + rel[row * 2 + 0] * W_ih[(size_t)orig * 1060 + 1058]
             + rel[row * 2 + 1] * W_ih[(size_t)orig * 1060 + 1059];
          gi[(size_t)row * NGI + c] = v;
        }
      }
    }
  }
}

// ---------------- recurrence: 24 GRU steps, h kept in LDS ----------------
// grid 256, block 256. Each block owns 64 batch rows; wave w owns h-cols [16w,16w+16)
// with its three gate fragments (r,z,n) via the 48-wide interleaved layout.
__global__ __launch_bounds__(256) void k_recur(
    const float* __restrict__ h0, const float* __restrict__ gi,
    const unsigned short* __restrict__ Whh, const float* __restrict__ Wa,
    const float* __restrict__ Wm, const float* __restrict__ a0g,
    const float* __restrict__ fut, const float* __restrict__ b_hh,
    const float* __restrict__ b_mu, const float* __restrict__ b_std,
    float* __restrict__ out) {
  __shared__ unsigned short h_sh[64][136];      // bf16 h tile, padded vs bank conflicts
  __shared__ float a_sh[64][2];
  __shared__ float wm_sh[4][128];

  int tid = threadIdx.x;
  int lane = tid & 31, wave = tid >> 5;
  int l15 = lane & 15, hi = lane >> 4;
  int rowbase = blockIdx.x * 64;
  int hcol = 16 * wave + l15;

  for (int i = tid; i < 4 * 128; i += 256) wm_sh[i >> 7][i & 127] = Wm[i];
  for (int i = tid; i < 64 * 128; i += 256) {
    int r = i >> 7, c = i & 127;
    h_sh[r][c] = f2bf(h0[(size_t)(rowbase + r) * H_ + c]);
  }

  float wa0[3], wa1[3], bhh[3];
  #pragma unroll
  for (int g = 0; g < 3; ++g) {
    int c = 48 * wave + 16 * g + l15;
    wa0[g] = Wa[c * 2 + 0];
    wa1[g] = Wa[c * 2 + 1];
    bhh[g] = b_hh[g * 128 + hcol];
  }
  float gir[4][3][8];                           // gi_base pinned in registers (const over T)
  #pragma unroll
  for (int mf = 0; mf < 4; ++mf)
    #pragma unroll
    for (int g = 0; g < 3; ++g) {
      int c = 48 * wave + 16 * g + l15;
      #pragma unroll
      for (int d = 0; d < 8; ++d)
        gir[mf][g][d] = gi[(size_t)(rowbase + 16 * mf + hi * 8 + d) * NGI + c];
    }
  float bm0 = b_mu[0], bm1 = b_mu[1], bs0 = b_std[0], bs1 = b_std[1];
  __syncthreads();

  for (int t = 0; t < T_SZ; ++t) {
    if (tid < 128) {                            // stage a_t (teacher forcing)
      int r = tid >> 1, j = tid & 1;
      a_sh[r][j] = (t == 0) ? a0g[(rowbase + r) * 2 + j]
                            : fut[((size_t)(t - 1) * B_SZ + rowbase + r) * 6 + 2 + j];
    }
    __syncthreads();

    v8f acc[4][3];
    #pragma unroll
    for (int mf = 0; mf < 4; ++mf)
      #pragma unroll
      for (int g = 0; g < 3; ++g)
        #pragma unroll
        for (int d = 0; d < 8; ++d) acc[mf][g][d] = bhh[g];   // gh bias seeds accum

    #pragma unroll
    for (int ks = 0; ks < 4; ++ks) {            // K = 128 = 4 x 32
      int kb = ks * 32;
      FragU bf[3];
      #pragma unroll
      for (int g = 0; g < 3; ++g) {
        const uint4* p = (const uint4*)(Whh + (size_t)(48 * wave + 16 * g + l15) * H_
                                            + kb + hi * 8);
        bf[g].q[0] = p[0]; bf[g].q[1] = p[2];
      }
      #pragma unroll
      for (int mf = 0; mf < 4; ++mf) {
        FragU af;                               // A fragment from LDS (ds_load_b128 x2)
        const uint4* p = (const uint4*)&h_sh[16 * mf + l15][kb + hi * 8];
        af.q[0] = p[0]; af.q[1] = p[2];
        #pragma unroll
        for (int g = 0; g < 3; ++g)
          acc[mf][g] = __builtin_amdgcn_wmma_f32_16x16x32_bf16(
              false, af.v, false, bf[g].v, (short)0, acc[mf][g], false, false);
      }
    }

    float hnew[4][8];
    #pragma unroll
    for (int mf = 0; mf < 4; ++mf) {
      #pragma unroll
      for (int d = 0; d < 8; ++d) {
        int rl = 16 * mf + hi * 8 + d;
        float av0 = a_sh[rl][0], av1 = a_sh[rl][1];
        float pr = gir[mf][0][d] + av0 * wa0[0] + av1 * wa1[0] + acc[mf][0][d];
        float pz = gir[mf][1][d] + av0 * wa0[1] + av1 * wa1[1] + acc[mf][1][d];
        float pn = gir[mf][2][d] + av0 * wa0[2] + av1 * wa1[2];
        float r  = fast_sigmoid(pr);
        float zg = fast_sigmoid(pz);
        float n  = fast_tanh(pn + r * acc[mf][2][d]);  // r multiplies gh_n (incl b_hh_n)
        hnew[mf][d] = (1.0f - zg) * n + zg * bf2f(h_sh[rl][hcol]);
      }
    }
    __syncthreads();                            // all reads of h done
    #pragma unroll
    for (int mf = 0; mf < 4; ++mf)
      #pragma unroll
      for (int d = 0; d < 8; ++d)
        h_sh[16 * mf + hi * 8 + d][hcol] = f2bf(hnew[mf][d]);
    __syncthreads();

    {                                           // mu/std heads: 1 (row,out) pair / thread
      int rl = tid >> 2, o = tid & 3;
      float dot = 0.0f;
      #pragma unroll 8
      for (int k = 0; k < 128; ++k) dot += bf2f(h_sh[rl][k]) * wm_sh[o][k];
      int rg = rowbase + rl;
      if (o < 2)
        out[(size_t)t * B_SZ * 2 + rg * 2 + o] = dot + (o ? bm1 : bm0);
      else
        out[(size_t)T_SZ * B_SZ * 2 + (size_t)t * B_SZ * 2 + rg * 2 + (o - 2)] =
            __expf(0.5f * (dot + ((o - 2) ? bs1 : bs0)));   // sqrt(exp(x)) = exp(x/2)
    }
  }
}

// ---------------- host launcher ----------------

extern "C" void kernel_launch(void* const* d_in, const int* in_sizes, int n_in,
                              void* d_out, int out_size, void* d_ws, size_t ws_size,
                              hipStream_t stream) {
  const float* last  = (const float*)d_in[0];
  const float* enc   = (const float*)d_in[1];
  const float* z     = (const float*)d_in[2];
  const float* sg    = (const float*)d_in[3];
  const float* fut   = (const float*)d_in[4];
  const float* W_dh  = (const float*)d_in[5];
  const float* b_dh  = (const float*)d_in[6];
  const float* W_vel = (const float*)d_in[7];
  const float* b_vel = (const float*)d_in[8];
  const float* W_ih  = (const float*)d_in[9];
  const float* b_ih  = (const float*)d_in[10];
  const float* W_hh  = (const float*)d_in[11];
  const float* b_hh  = (const float*)d_in[12];
  const float* W_mu  = (const float*)d_in[13];
  const float* b_mu  = (const float*)d_in[14];
  const float* W_std = (const float*)d_in[15];
  const float* b_std = (const float*)d_in[16];
  float* out = (float*)d_out;

  char* ws = (char*)d_ws;
  size_t off = 0;
  auto take = [&](size_t bytes) {
    char* p = ws + off;
    off = (off + bytes + 255) & ~(size_t)255;
    return p;
  };
  unsigned short* zx_bf = (unsigned short*)take((size_t)B_SZ * K1 * 2);
  unsigned short* Wcat  = (unsigned short*)take((size_t)512 * K1 * 2);
  unsigned short* Whh   = (unsigned short*)take((size_t)NGI * H_ * 2);
  float* Wa  = (float*)take((size_t)NGI * 2 * 4);
  float* Wm  = (float*)take((size_t)4 * H_ * 4);
  float* rel = (float*)take((size_t)B_SZ * 2 * 4);
  float* a0  = (float*)take((size_t)B_SZ * 2 * 4);
  float* h0  = (float*)take((size_t)B_SZ * H_ * 4);
  float* gi  = (float*)take((size_t)B_SZ * NGI * 4);

  { int total = B_SZ * K1;
    k_conv_zx<<<(total + 255) / 256, 256, 0, stream>>>(enc, z, zx_bf, total); }
  { int total = 512 * K1 + NGI * H_ + NGI * 2 + 4 * H_;
    k_conv_w<<<(total + 255) / 256, 256, 0, stream>>>(W_dh, W_ih, W_hh, W_mu, W_std,
                                                      Wcat, Whh, Wa, Wm); }
  k_conv_ra<<<(B_SZ + 255) / 256, 256, 0, stream>>>(last, sg, W_vel, b_vel, rel, a0);

  dim3 g(B_SZ / 128, 4);
  k_gemm_pre<<<g, 256, 0, stream>>>(zx_bf, Wcat, b_dh, b_ih, W_ih, rel, h0, gi);
  k_recur<<<B_SZ / 64, 256, 0, stream>>>(h0, gi, Whh, Wa, Wm, a0, fut,
                                         b_hh, b_mu, b_std, out);
}
